// CorrelationVolume_14963666059406
// MI455X (gfx1250) — matrix operand order repaired and checked by reading, not compiled, not run on hardware
//
#include <hip/hip_runtime.h>
#include <stdint.h>

typedef __attribute__((ext_vector_type(16))) _Float16 v16h;
typedef __attribute__((ext_vector_type(8)))  _Float16 v8h;
typedef __attribute__((ext_vector_type(8)))  float    v8f;
typedef __attribute__((ext_vector_type(4)))  int      v4i;

#define NIMG 98
#define FH 64
#define FW 96
#define FHW (FH*FW)

#if __has_builtin(__builtin_amdgcn_global_load_async_to_lds_b128)
#define HAVE_ASYNC_LDS 1
#else
#define HAVE_ASYNC_LDS 0
#endif

typedef __attribute__((address_space(1))) v4i as1_v4i;   // global int4
typedef __attribute__((address_space(3))) v4i as3_v4i;   // LDS int4

// ---------------------------------------------------------------------------
// 1) Correlation volume: 98 images, NHWC f16 [img][y][x][64ch]
//    ch 0..31 = fmap1*mask*valid ; ch 32..63 = shifted(fmap2)*valid
// ---------------------------------------------------------------------------
__global__ void build_volume_kernel(const float* __restrict__ f1,
                                    const float* __restrict__ f2,
                                    _Float16* __restrict__ out)
{
    int gid = blockIdx.x * blockDim.x + threadIdx.x;
    if (gid >= NIMG * FHW) return;
    int img = gid / FHW;
    int p   = gid % FHW;
    int y = p / FW, x = p % FW;
    int bb = img / 49;
    int ij = img % 49;
    int i = ij / 7, j = ij % 7;
    int di = i - 3, dj = j - 3;          // di shifts W, dj shifts H (per reference)
    int sy = y + dj, sx = x + di;
    bool inr = (sy >= 0) && (sy < FH) && (sx >= 0) && (sx < FW);

    const float* f1p = f1 + (size_t)bb * 32 * FHW + p;
    const float* f2p = f2 + (size_t)bb * 32 * FHW + (inr ? (sy * FW + sx) : 0);

    float sh[32];
    float s = 0.f;
    #pragma unroll 4
    for (int c = 0; c < 32; ++c) {
        float v = inr ? f2p[(size_t)c * FHW] : 0.f;
        sh[c] = v;
        s += v;
    }
    float valid = (s != 0.f) ? 1.f : 0.f;
    float m1 = inr ? valid : 0.f;

    _Float16* op = out + (size_t)gid * 64;
    #pragma unroll 4
    for (int c = 0; c < 32; ++c) op[c]      = (_Float16)(f1p[(size_t)c * FHW] * m1);
    #pragma unroll 4
    for (int c = 0; c < 32; ++c) op[32 + c] = (_Float16)(sh[c] * valid);
}

// ---------------------------------------------------------------------------
// 2) Pack conv weights (f32 OIHW, or IOHW for deconv) into per-lane A-fragment
//    order for V_WMMA_F32_16X16X32_F16 (this file's contiguous k, not the ISA table order):
//    pw[(((t*KC+kc)*OCT+oct)*32 + lane)*16 + hi]
//    lane<16 : M=lane,    K=hi
//    lane>=16: M=lane-16, K=16+hi
// ---------------------------------------------------------------------------
__global__ void pack_weights_kernel(const float* __restrict__ w,
                                    _Float16* __restrict__ pw,
                                    int OC, int IC, int taps, int OCT, int isDeconv)
{
    int idx = blockIdx.x * blockDim.x + threadIdx.x;
    int KC = IC / 32;
    int total = taps * KC * OCT * 32 * 16;
    if (idx >= total) return;
    int hi   = idx & 15;
    int lane = (idx >> 4) & 31;
    int rest = idx >> 9;
    int oct  = rest % OCT; rest /= OCT;
    int kc   = rest % KC;
    int t    = rest / KC;
    int M = lane & 15;
    int K = (lane < 16) ? hi : (16 + hi);
    int oc = oct * 16 + M;
    int ic = kc * 32 + K;
    float v = 0.f;
    if (oc < OC && ic < IC)
        v = isDeconv ? w[((size_t)ic * OC + oc) * taps + t]
                     : w[((size_t)oc * IC + ic) * taps + t];
    pw[idx] = (_Float16)v;
}

// ---------------------------------------------------------------------------
// 3) Implicit-GEMM conv / transposed-conv via v_wmma_f32_16x16x32_f16.
//    Fully specialized; 4 waves/block, each wave = one 16oc x 16pixel tile.
//    Weight A-fragments for this block's oc-tile are staged in LDS via
//    async global->LDS loads (s_wait_asynccnt), then read with ds_load_b128.
//    act: f16 NHWC [img][IH][IW][IC] ; out: f32 NCHW [img][OC][OH*OW]
// ---------------------------------------------------------------------------
template<int IH, int IW, int IC, int OH, int OW, int OC,
         int STRIDE, int TAPS, int KD, bool DEC>
__global__ __launch_bounds__(128) void conv_wmma_kernel(
    const _Float16* __restrict__ act, const _Float16* __restrict__ pw,
    float* __restrict__ out)
{
    constexpr int KC  = IC / 32;
    constexpr int OCT = OC / 16;
    constexpr int OHW = OH * OW;

    const int lane = threadIdx.x & 31;
    const int wave = threadIdx.x >> 5;
    const int tile = blockIdx.x * 4 + wave;       // 16-pixel tile id
    const int oct  = blockIdx.y;                  // oc tile (16 ocs)
    const int img  = blockIdx.z;

    // --- stage this oc-tile's weight fragments into LDS ---
    __shared__ _Float16 wlds[TAPS * KC * 512];
    {
        constexpr int units = TAPS * KC * 64;     // 16-byte units
        for (int u = threadIdx.x; u < units; u += 128) {
            const int b   = u >> 6;               // (t*KC+kc) block
            const int sub = u & 63;
            const _Float16* g = pw + (((size_t)b * OCT + oct) * 512 + sub * 8);
            _Float16*       l = wlds + (b * 512 + sub * 8);
#if HAVE_ASYNC_LDS
            __builtin_amdgcn_global_load_async_to_lds_b128(
                (as1_v4i*)(uintptr_t)g, (as3_v4i*)(uint32_t)(uintptr_t)l, 0, 0);
#else
            *(v8h*)l = *(const v8h*)g;
#endif
        }
#if HAVE_ASYNC_LDS
        asm volatile("s_wait_asynccnt 0" ::: "memory");
#endif
        __syncthreads();
    }

    const int p  = tile * 16 + (lane & 15);       // output pixel (tile fits in one row)
    const int oy = p / OW;
    const int ox = p % OW;
    const int coff = (lane < 16) ? 0 : 16;        // B-fragment K-half select

    const _Float16* actImg = act + (size_t)img * IH * IW * IC;

    v8f acc = {};
    #pragma unroll
    for (int t = 0; t < TAPS; ++t) {
        constexpr int pad = 1;
        const int ky = t / KD, kx = t % KD;
        int iy, ix; bool inb;
        if (!DEC) {
            iy  = oy * STRIDE + ky - pad;
            ix  = ox * STRIDE + kx - pad;
            inb = (iy >= 0) && (iy < IH) && (ix >= 0) && (ix < IW);
        } else {        // transposed conv, stride 2, pad 1, k=4 (gather form)
            int ty = oy + 1 - ky, tx = ox + 1 - kx;
            iy  = ty >> 1;  ix = tx >> 1;
            inb = ((ty & 1) == 0) && ((tx & 1) == 0) &&
                  (ty >= 0) && (tx >= 0) && (iy < IH) && (ix < IW);
        }
        const _Float16* bp = actImg + ((size_t)iy * IW + ix) * IC + coff;
        #pragma unroll
        for (int kc = 0; kc < KC; ++kc) {
            v16h afrag = *(const v16h*)(wlds + (((t * KC + kc) * 32) + lane) * 16);
            v16h bfrag = {};
            if (inb) bfrag = *(const v16h*)(bp + kc * 32);
            acc = __builtin_amdgcn_wmma_f32_16x16x32_f16(
                      false, afrag, false, bfrag, (short)0, acc, false, false);
        }
    }

    // C layout: VGPR r -> M=r (lanes 0-15) / M=8+r (lanes 16-31), N=lane&15
    const int mbase = (lane < 16) ? 0 : 8;
    float* op = out + ((size_t)img * OC + oct * 16 + mbase) * OHW + p;
    #pragma unroll
    for (int r = 0; r < 8; ++r)
        op[(size_t)r * OHW] = acc[r];
}

// ---------------------------------------------------------------------------
// 4) Instance-norm statistics: one block per (img,channel) plane.
// ---------------------------------------------------------------------------
__global__ __launch_bounds__(256) void stats_kernel(const float* __restrict__ x,
                                                    float* __restrict__ stats, int HW)
{
    __shared__ float ssum[256], ssq[256];
    const float* xp = x + (size_t)blockIdx.x * HW;
    float s = 0.f, q = 0.f;
    for (int i = threadIdx.x; i < HW; i += 256) {
        float v = xp[i];
        s += v; q += v * v;
    }
    ssum[threadIdx.x] = s; ssq[threadIdx.x] = q;
    __syncthreads();
    for (int off = 128; off > 0; off >>= 1) {
        if (threadIdx.x < off) {
            ssum[threadIdx.x] += ssum[threadIdx.x + off];
            ssq[threadIdx.x]  += ssq[threadIdx.x + off];
        }
        __syncthreads();
    }
    if (threadIdx.x == 0) {
        float inv  = 1.f / (float)HW;
        float mean = ssum[0] * inv;
        float var  = ssq[0] * inv - mean * mean;
        stats[(size_t)blockIdx.x * 2 + 0] = mean;
        stats[(size_t)blockIdx.x * 2 + 1] = rsqrtf(var + 1e-5f);
    }
}

// ---------------------------------------------------------------------------
// 5) Normalize + leaky ReLU + repack f32 NCHW -> f16 NHWC for next layer.
// ---------------------------------------------------------------------------
__global__ void norm_leaky_pack_kernel(const float* __restrict__ x,
                                       const float* __restrict__ stats,
                                       _Float16* __restrict__ act, int OC, int HW)
{
    size_t gid = (size_t)blockIdx.x * blockDim.x + threadIdx.x;
    size_t total = (size_t)NIMG * OC * HW;
    if (gid >= total) return;
    int p    = (int)(gid % HW);
    size_t r = gid / HW;
    int oc   = (int)(r % OC);
    int img  = (int)(r / OC);
    size_t g = (size_t)img * OC + oc;
    float v = (x[gid] - stats[g * 2]) * stats[g * 2 + 1];
    v = (v >= 0.f) ? v : 0.01f * v;
    act[((size_t)img * HW + p) * OC + oc] = (_Float16)v;
}

// ---------------------------------------------------------------------------
// 6) Final 3x3 conv, 32 -> 1 channel, + bias. Output (b,7,7,h,w) == img-major.
// ---------------------------------------------------------------------------
__global__ void conv6_kernel(const _Float16* __restrict__ act,
                             const float* __restrict__ w6,
                             const float* __restrict__ b6,
                             float* __restrict__ out)
{
    int gid = blockIdx.x * blockDim.x + threadIdx.x;
    if (gid >= NIMG * FHW) return;
    int img = gid / FHW, p = gid % FHW;
    int y = p / FW, x = p % FW;
    float s = b6[0];
    for (int ky = 0; ky < 3; ++ky) {
        int iy = y + ky - 1;
        if (iy < 0 || iy >= FH) continue;
        for (int kx = 0; kx < 3; ++kx) {
            int ix = x + kx - 1;
            if (ix < 0 || ix >= FW) continue;
            const _Float16* ap = act + ((size_t)img * FHW + iy * FW + ix) * 32;
            const float* wp = w6 + (ky * 3 + kx);
            #pragma unroll 4
            for (int c = 0; c < 32; ++c)
                s += (float)ap[c] * wp[(size_t)c * 9];
        }
    }
    out[gid] = s;
}

// ---------------------------------------------------------------------------
extern "C" void kernel_launch(void* const* d_in, const int* in_sizes, int n_in,
                              void* d_out, int out_size, void* d_ws, size_t ws_size,
                              hipStream_t stream) {
    const float* fmap1 = (const float*)d_in[0];
    const float* fmap2 = (const float*)d_in[1];
    const float* w1 = (const float*)d_in[2];
    const float* w2 = (const float*)d_in[3];
    const float* w3 = (const float*)d_in[4];
    const float* w4 = (const float*)d_in[5];
    const float* w5 = (const float*)d_in[6];
    const float* w6 = (const float*)d_in[7];
    const float* b6 = (const float*)d_in[8];
    float* outp = (float*)d_out;

    char* ws = (char*)d_ws;
    size_t off = 0;
    auto carve = [&](size_t bytes) -> void* {
        void* p = ws + off;
        off = (off + bytes + 255) & ~(size_t)255;
        return p;
    };

    _Float16* b16a  = (_Float16*)carve((size_t)NIMG * FHW * 64 * 2);  // mvol / act2 / act4
    _Float16* b16b  = (_Float16*)carve((size_t)NIMG * FHW * 96 * 2);  // act1 / act3 / act5
    float*    b32   = (float*)   carve((size_t)NIMG * FHW * 96 * 4);  // conv f32 ping buffer
    _Float16* pw1   = (_Float16*)carve(512 << 10);
    _Float16* pw2   = (_Float16*)carve(512 << 10);
    _Float16* pw3   = (_Float16*)carve(512 << 10);
    _Float16* pw4   = (_Float16*)carve(512 << 10);
    _Float16* pw5   = (_Float16*)carve(512 << 10);
    float*    stats = (float*)   carve((size_t)NIMG * 128 * 2 * 4);

    // --- correlation volume -> f16 NHWC ---
    {
        int n = NIMG * FHW;
        build_volume_kernel<<<(n + 255) / 256, 256, 0, stream>>>(fmap1, fmap2, b16a);
    }

    // --- pack weights into WMMA A-fragment order ---
    auto pack = [&](const float* w, _Float16* pw, int OC, int IC, int taps, int OCT, int dec) {
        int total = taps * (IC / 32) * OCT * 32 * 16;
        pack_weights_kernel<<<(total + 255) / 256, 256, 0, stream>>>(w, pw, OC, IC, taps, OCT, dec);
    };
    pack(w1, pw1,  96,  64,  9, 6, 0);
    pack(w2, pw2, 128,  96,  9, 8, 0);
    pack(w3, pw3, 128, 128,  9, 8, 0);
    pack(w4, pw4,  64, 128,  9, 4, 0);
    pack(w5, pw5,  32,  64, 16, 2, 1);   // (I,O,4,4) -> gather-form deconv weights

    auto post = [&](int OC, int OHW, _Float16* act_out) {
        stats_kernel<<<NIMG * OC, 256, 0, stream>>>(b32, stats, OHW);
        size_t total = (size_t)NIMG * OC * OHW;
        norm_leaky_pack_kernel<<<(unsigned)((total + 255) / 256), 256, 0, stream>>>(
            b32, stats, act_out, OC, OHW);
    };

    // conv1: 64ch 64x96 -> 96ch 64x96
    conv_wmma_kernel<64,96, 64, 64,96, 96, 1, 9,3,false>
        <<<dim3(96, 6, NIMG), 128, 0, stream>>>(b16a, pw1, b32);
    post(96, 6144, b16b);

    // conv2 (stride 2): 96ch 64x96 -> 128ch 32x48
    conv_wmma_kernel<64,96, 96, 32,48,128, 2, 9,3,false>
        <<<dim3(24, 8, NIMG), 128, 0, stream>>>(b16b, pw2, b32);
    post(128, 1536, b16a);

    // conv3: 128ch 32x48 -> 128ch 32x48
    conv_wmma_kernel<32,48,128, 32,48,128, 1, 9,3,false>
        <<<dim3(24, 8, NIMG), 128, 0, stream>>>(b16a, pw3, b32);
    post(128, 1536, b16b);

    // conv4: 128ch 32x48 -> 64ch 32x48
    conv_wmma_kernel<32,48,128, 32,48, 64, 1, 9,3,false>
        <<<dim3(24, 4, NIMG), 128, 0, stream>>>(b16b, pw4, b32);
    post(64, 1536, b16a);

    // deconv5 (transposed, stride 2, k=4): 64ch 32x48 -> 32ch 64x96
    conv_wmma_kernel<32,48, 64, 64,96, 32, 2,16,4,true>
        <<<dim3(96, 2, NIMG), 128, 0, stream>>>(b16a, pw5, b32);
    post(32, 6144, b16b);

    // --- final 1-channel conv + bias -> d_out ---
    {
        int n = NIMG * FHW;
        conv6_kernel<<<(n + 255) / 256, 256, 0, stream>>>(b16b, w6, b6, outp);
    }
}
